// TorchLMHeadGRPO_27539330302083
// MI455X (gfx1250) — matrix operand order repaired and checked by reading, not compiled, not run on hardware
//
#include <hip/hip_runtime.h>
#include <hip/hip_bf16.h>
#include <math.h>

// ---------------- problem constants (reference shapes) ----------------
constexpr int   Bc = 2, Sc = 1024, Hc = 2048, Vc = 32000;
constexpr float BETA = 0.1f;
constexpr int   Mrows = Bc * Sc;                 // 2048 rows per GEMM
constexpr int   KSTEP = 32;                      // bf16 WMMA K
constexpr int   MT = 2;                          // m-tiles (16 rows each) per wave
constexpr int   NT = 4;                          // n-tiles (16 cols each) per pass
constexpr int   ROWS_PER_WAVE = 16 * MT;         // 32
constexpr int   PASS_COLS = 16 * NT;             // 64
constexpr int   SLICES = 25;
constexpr int   SLICE_COLS = Vc / SLICES;        // 1280
constexpr int   PASSES = SLICE_COLS / PASS_COLS; // 20
constexpr int   ROWBLOCKS = Mrows / ROWS_PER_WAVE;   // 64
constexpr int   WAVES_PER_GEMM = ROWBLOCKS * SLICES; // 1600
constexpr int   WAVES_PER_BLOCK = 8;             // 256 threads, wave32

typedef __bf16 bf16_t;
typedef bf16_t v16bf __attribute__((ext_vector_type(16)));
typedef bf16_t v8bf  __attribute__((ext_vector_type(8)));
typedef float  v8f   __attribute__((ext_vector_type(8)));

union V16U { v16bf v; v8bf h[2]; };

// ---------------- stage 1: fp32 -> bf16 (RNE), 8 elems / thread ----------------
__device__ __forceinline__ unsigned int f2bf_pk(float lo, float hi) {
    unsigned int ul = __float_as_uint(lo);
    unsigned int uh = __float_as_uint(hi);
    ul += 0x7fffu + ((ul >> 16) & 1u);
    uh += 0x7fffu + ((uh >> 16) & 1u);
    return (ul >> 16) | (uh & 0xffff0000u);
}

__global__ __launch_bounds__(256)
void cvt_f32_to_bf16_x8(const float* __restrict__ src,
                        unsigned short* __restrict__ dst, long n8) {
    long i = (long)blockIdx.x * blockDim.x + threadIdx.x;
    if (i >= n8) return;
    const float4* s = reinterpret_cast<const float4*>(src) + i * 2;
    float4 f0 = s[0];
    float4 f1 = s[1];
    uint4 o;
    o.x = f2bf_pk(f0.x, f0.y);
    o.y = f2bf_pk(f0.z, f0.w);
    o.z = f2bf_pk(f1.x, f1.y);
    o.w = f2bf_pk(f1.z, f1.w);
    reinterpret_cast<uint4*>(dst)[i] = o;
}

// ---------------- stage 2: fused GEMM (bf16 WMMA) + online softmax ----------------
// One wave handles: 32 rows (2 m-tiles) x one N-slice of 1280 columns, K fully reduced.
// Each B fragment feeds 2 WMMAs (M-reuse) to stay under the WGP memory path.
// A layout (ISA 16-bit A table): lane<16 -> M=lane, K in {k0..k0+7, k0+16..k0+23}, k0=0;
//                                lane>=16 -> same M set, k0=8.
// B layout (same as A): lane<16 -> N=lane, K in {k0..k0+7, k0+16..k0+23}, k0=0;
//                       lane>=16 -> k0=8.
// C/D layout: element j of lane L holds row M = j + 8*(L/16), col N = ntile + (L%16).
__global__ __launch_bounds__(256)
void lmhead_gemm_softmax(const unsigned short* __restrict__ xbf,
                         const unsigned short* __restrict__ rxbf,
                         const unsigned short* __restrict__ wbf,
                         const unsigned short* __restrict__ rwbf,
                         float* __restrict__ partials) // [2][Mrows][SLICES][3]
{
    const int wid  = blockIdx.x * WAVES_PER_BLOCK + (threadIdx.x >> 5);
    const int lane = threadIdx.x & 31;
    const int gemm = wid / WAVES_PER_GEMM;
    const int rem  = wid % WAVES_PER_GEMM;
    const int rowblock = rem / SLICES;
    const int slice    = rem % SLICES;

    const bf16_t* __restrict__ A = reinterpret_cast<const bf16_t*>(gemm ? rxbf : xbf);
    const bf16_t* __restrict__ W = reinterpret_cast<const bf16_t*>(gemm ? rwbf : wbf);

    const int  l16     = lane & 15;
    const int  half    = lane >> 4;
    const int  ka_half = half * 8;   // A: 0 or 8
    const int  kb_half = half * 8;   // B: 0 or 8
    const long arow0   = (long)(rowblock * ROWS_PER_WAVE + l16) * Hc;
    const long arow1   = arow0 + (long)16 * Hc;

    // per-lane online softmax partials: row = mt*16 + j + 8*half
    float mx[MT][8], se[MT][8], sm[MT][8];
#pragma unroll
    for (int mt = 0; mt < MT; ++mt)
#pragma unroll
        for (int j = 0; j < 8; ++j) { mx[mt][j] = -INFINITY; se[mt][j] = 0.f; sm[mt][j] = 0.f; }

    const int n0 = slice * SLICE_COLS;
    for (int p = 0; p < PASSES; ++p) {
        const int nb = n0 + p * PASS_COLS;
        v8f acc[MT][NT] = {};
        const bf16_t* __restrict__ ap0 = A + arow0 + ka_half;
        const bf16_t* __restrict__ ap1 = A + arow1 + ka_half;
        const bf16_t* __restrict__ wp0 = W + (long)(nb + l16) * Hc + kb_half;

        for (int k = 0; k < Hc; k += KSTEP) {
            V16U a0, a1;
            a0.h[0] = *reinterpret_cast<const v8bf*>(ap0 + k);
            a0.h[1] = *reinterpret_cast<const v8bf*>(ap0 + k + 16);
            a1.h[0] = *reinterpret_cast<const v8bf*>(ap1 + k);
            a1.h[1] = *reinterpret_cast<const v8bf*>(ap1 + k + 16);
#pragma unroll
            for (int t = 0; t < NT; ++t) {
                const bf16_t* wp = wp0 + (long)t * 16 * Hc + k;
                V16U b;
                b.h[0] = *reinterpret_cast<const v8bf*>(wp);
                b.h[1] = *reinterpret_cast<const v8bf*>(wp + 16);
                acc[0][t] = __builtin_amdgcn_wmma_f32_16x16x32_bf16(
                    false, a0.v, false, b.v, (short)0, acc[0][t], false, false);
                acc[1][t] = __builtin_amdgcn_wmma_f32_16x16x32_bf16(
                    false, a1.v, false, b.v, (short)0, acc[1][t], false, false);
            }
        }
        // fold this pass's 64 columns into per-lane running stats
#pragma unroll
        for (int mt = 0; mt < MT; ++mt) {
#pragma unroll
            for (int t = 0; t < NT; ++t) {
#pragma unroll
                for (int j = 0; j < 8; ++j) {
                    float v  = acc[mt][t][j];
                    float nm = fmaxf(mx[mt][j], v);
                    se[mt][j] = se[mt][j] * __expf(mx[mt][j] - nm) + __expf(v - nm);
                    mx[mt][j] = nm;
                    sm[mt][j] += v;
                }
            }
        }
    }

    // merge across the 16 lanes that share each row (xor strides stay in-half)
#pragma unroll
    for (int d = 1; d < 16; d <<= 1) {
#pragma unroll
        for (int mt = 0; mt < MT; ++mt) {
#pragma unroll
            for (int j = 0; j < 8; ++j) {
                float om = __shfl_xor(mx[mt][j], d, 32);
                float os = __shfl_xor(se[mt][j], d, 32);
                float ot = __shfl_xor(sm[mt][j], d, 32);
                float nm = fmaxf(mx[mt][j], om);
                se[mt][j] = se[mt][j] * __expf(mx[mt][j] - nm) + os * __expf(om - nm);
                mx[mt][j] = nm;
                sm[mt][j] += ot;
            }
        }
    }

    if (l16 == 0) {
#pragma unroll
        for (int mt = 0; mt < MT; ++mt) {
            const int rbase = rowblock * ROWS_PER_WAVE + mt * 16 + half * 8;
#pragma unroll
            for (int j = 0; j < 8; ++j) {
                long idx = (((long)gemm * Mrows + rbase + j) * SLICES + slice) * 3;
                partials[idx + 0] = mx[mt][j];
                partials[idx + 1] = se[mt][j];
                partials[idx + 2] = sm[mt][j];
            }
        }
    }
}

// ---------------- stage 3: per-row slice merge + masked per-(gemm,b) sums ----------------
__global__ __launch_bounds__(256)
void combine_rows(const float* __restrict__ partials,
                  const int* __restrict__ amask,   // [B,S]
                  float* __restrict__ stats)       // [2][B][2] = {masked logp sum, logits row-sum}
{
    const int g = blockIdx.x >> 1;
    const int b = blockIdx.x & 1;
    const int tid = threadIdx.x;

    float lsum = 0.f, vsum = 0.f;
    for (int s = tid; s < Sc; s += 256) {
        const int row = b * Sc + s;
        const float* P = partials + ((long)g * Mrows + row) * SLICES * 3;
        float m = -INFINITY, S = 0.f, tot = 0.f;
        for (int sl = 0; sl < SLICES; ++sl) {
            float pm = P[sl * 3 + 0];
            float ps = P[sl * 3 + 1];
            float nm = fmaxf(m, pm);
            S = S * __expf(m - nm) + ps * __expf(pm - nm);
            m = nm;
            tot += P[sl * 3 + 2];
        }
        float logp = -logf(S);        // m - lse == -log(sum exp(l - m_global))
        if (amask[row] != 0) lsum += logp;
        vsum += tot;
    }

    __shared__ float sh0[256], sh1[256];
    sh0[tid] = lsum; sh1[tid] = vsum;
    __syncthreads();
    for (int o = 128; o > 0; o >>= 1) {
        if (tid < o) { sh0[tid] += sh0[tid + o]; sh1[tid] += sh1[tid + o]; }
        __syncthreads();
    }
    if (tid == 0) {
        stats[(g * Bc + b) * 2 + 0] = sh0[0];
        stats[(g * Bc + b) * 2 + 1] = sh1[0];
    }
}

// ---------------- stage 4: scalar finalize -> 5 outputs ----------------
__global__ void finalize(const float* __restrict__ stats,
                         const float* __restrict__ rewards,
                         float* __restrict__ out)
{
    if (threadIdx.x != 0 || blockIdx.x != 0) return;
    float p[Bc], q[Bc], r[Bc];
    float logits_sum = 0.f;
    for (int b = 0; b < Bc; ++b) {
        p[b] = stats[(0 * Bc + b) * 2 + 0];
        q[b] = stats[(1 * Bc + b) * 2 + 0];
        logits_sum += stats[(0 * Bc + b) * 2 + 1];
        r[b] = rewards[b];
    }
    float rm = 0.f;
    for (int b = 0; b < Bc; ++b) rm += r[b];
    rm /= (float)Bc;
    float rv = 0.f;
    for (int b = 0; b < Bc; ++b) rv += (r[b] - rm) * (r[b] - rm);
    float rstd = sqrtf(rv / (float)(Bc - 1));

    float loss = 0.f, pmean = 0.f, klmean = 0.f;
    for (int b = 0; b < Bc; ++b) {
        float adv = r[b] - rm;
        if (rstd > 0.f) adv /= rstd;
        float kl = p[b] - q[b];
        loss   += -adv * p[b] + BETA * kl;
        pmean  += p[b];
        klmean += kl;
    }
    loss /= (float)Bc; pmean /= (float)Bc; klmean /= (float)Bc;
    float pv = 0.f;
    for (int b = 0; b < Bc; ++b) pv += (p[b] - pmean) * (p[b] - pmean);
    float pstd = sqrtf(pv / (float)(Bc - 1));

    out[0] = loss;
    out[1] = pmean;
    out[2] = pstd;
    out[3] = logits_sum / ((float)Bc * (float)Sc * (float)Vc);
    out[4] = klmean;
}

// ---------------- host launcher ----------------
extern "C" void kernel_launch(void* const* d_in, const int* in_sizes, int n_in,
                              void* d_out, int out_size, void* d_ws, size_t ws_size,
                              hipStream_t stream) {
    const float* x       = (const float*)d_in[0];
    const float* w       = (const float*)d_in[1];
    const int*   amask   = (const int*)d_in[2];
    const float* rewards = (const float*)d_in[3];
    const float* rx      = (const float*)d_in[4];
    const float* rw      = (const float*)d_in[5];

    const size_t nX = (size_t)Mrows * Hc;   // 4,194,304
    const size_t nW = (size_t)Vc * Hc;      // 65,536,000

    // workspace layout (~282 MB): bf16 copies, per-row/slice partials, 8 stat floats
    unsigned short* xbf  = (unsigned short*)d_ws;
    unsigned short* rxbf = xbf  + nX;
    unsigned short* wbf  = rxbf + nX;
    unsigned short* rwbf = wbf  + nW;
    float* partials = (float*)(rwbf + nW);                       // 2*2048*25*3 floats
    float* stats    = partials + (size_t)2 * Mrows * SLICES * 3; // 8 floats

    // stage 1: convert all four fp32 tensors to bf16
    cvt_f32_to_bf16_x8<<<(int)(nX / 8 / 256), 256, 0, stream>>>(x,  xbf,  (long)(nX / 8));
    cvt_f32_to_bf16_x8<<<(int)(nX / 8 / 256), 256, 0, stream>>>(rx, rxbf, (long)(nX / 8));
    cvt_f32_to_bf16_x8<<<(int)(nW / 8 / 256), 256, 0, stream>>>(w,  wbf,  (long)(nW / 8));
    cvt_f32_to_bf16_x8<<<(int)(nW / 8 / 256), 256, 0, stream>>>(rw, rwbf, (long)(nW / 8));

    // stage 2: both GEMMs + online softmax partials (3200 waves, 8 waves/block)
    lmhead_gemm_softmax<<<(2 * WAVES_PER_GEMM) / WAVES_PER_BLOCK, 256, 0, stream>>>(
        xbf, rxbf, wbf, rwbf, partials);

    // stage 3: per-row merge + masked reductions (one block per (gemm, b))
    combine_rows<<<4, 256, 0, stream>>>(partials, amask, stats);

    // stage 4: scalars
    finalize<<<1, 32, 0, stream>>>(stats, rewards, (float*)d_out);
}
